// SAGE_48730698940921
// MI455X (gfx1250) — matrix operand order repaired and checked by reading, not compiled, not run on hardware
//
#include <hip/hip_runtime.h>

typedef unsigned short u16;
typedef __attribute__((ext_vector_type(16))) __bf16 v16bf;
typedef __attribute__((ext_vector_type(8)))  float  v8f;

#define N_NODES 50000
#define N_EDGES 800000
#define D_IN    128
#define D_HID   256
#define D_OUT   47
#define BN_EPS  1e-5f

// ---------- bf16 helpers (RNE) ----------
__device__ __forceinline__ u16 f2bf(float f) {
  union { float f; unsigned u; } c; c.f = f;
  unsigned u = c.u + 0x7FFFu + ((c.u >> 16) & 1u);
  return (u16)(u >> 16);
}
__device__ __forceinline__ float bf2f(u16 h) {
  union { unsigned u; float f; } c; c.u = ((unsigned)h) << 16;
  return c.f;
}
__device__ __forceinline__ unsigned pk_scale(unsigned p, float sc) {
  return (unsigned)f2bf(bf2f((u16)(p & 0xFFFFu)) * sc) |
         ((unsigned)f2bf(bf2f((u16)(p >> 16)) * sc) << 16);
}

// CDNA5 async global -> LDS copy (8 bytes per lane, tracked on ASYNCcnt)
__device__ __forceinline__ void async_b64(const u16* g, unsigned lds_off) {
  asm volatile("global_load_async_to_lds_b64 %0, %1, off"
               :: "v"(lds_off), "v"((unsigned long long)g) : "memory");
}
__device__ __forceinline__ void wait_async0() {
  asm volatile("s_wait_asynccnt 0x0" ::: "memory");
}

// ---------------- utility kernels ----------------

__global__ void k_zero_u32(unsigned* __restrict__ p, int n) {
  int i = blockIdx.x * blockDim.x + threadIdx.x;
  if (i < n) p[i] = 0u;
}

__global__ void k_degree(const int* __restrict__ ei, float* __restrict__ deg) {
  int e = blockIdx.x * blockDim.x + threadIdx.x;
  if (e < N_EDGES) atomicAdd(&deg[ei[N_EDGES + e]], 1.0f);
}

__global__ void k_invdeg(float* __restrict__ deg) {
  int i = blockIdx.x * blockDim.x + threadIdx.x;
  if (i < N_NODES) deg[i] = 1.0f / fmaxf(deg[i], 1.0f);
}

// f32 -> bf16, 4 elements per thread
__global__ void k_cvt_bf16_v4(const float* __restrict__ s, u16* __restrict__ d, int n4) {
  int i = blockIdx.x * blockDim.x + threadIdx.x;
  if (i >= n4) return;
  float4 v = ((const float4*)s)[i];
  uint2 o;
  o.x = (unsigned)f2bf(v.x) | ((unsigned)f2bf(v.y) << 16);
  o.y = (unsigned)f2bf(v.z) | ((unsigned)f2bf(v.w) << 16);
  ((uint2*)d)[i] = o;
}

// W[K][Ncols] f32 -> Wt[NPad][K] bf16 (transposed + zero-padded cols)
__global__ void k_wt_bf16(const float* __restrict__ W, u16* __restrict__ Wt,
                          int kshift, int Ncols, int total) {
  int idx = blockIdx.x * blockDim.x + threadIdx.x;
  if (idx >= total) return;
  int n = idx >> kshift;
  int k = idx & ((1 << kshift) - 1);
  float v = (n < Ncols) ? W[k * Ncols + n] : 0.0f;
  Wt[idx] = f2bf(v);
}

// bf16 scatter-add with fused inv_deg[dst] pre-scale:
//   agg[dst] = sum(feat[src] * inv_deg[dst]) == inv_deg[dst] * sum(feat[src])
// 4 channels per thread -> two packed-bf16 global atomics.
__global__ void k_scatter_bf16(const u16* __restrict__ feat, const int* __restrict__ ei,
                               const float* __restrict__ inv_deg,
                               u16* __restrict__ agg, int D, int qshift) {
  int tid = threadIdx.x;
  int e = blockIdx.x * (256 >> qshift) + (tid >> qshift);
  int c = (tid & ((1 << qshift) - 1)) << 2;
  int src = ei[e];
  int dst = ei[N_EDGES + e];
  float sc = inv_deg[dst];
  uint2 d = *(const uint2*)(feat + src * D + c);
  d.x = pk_scale(d.x, sc);
  d.y = pk_scale(d.y, sc);
  unsigned long long addr = (unsigned long long)(agg + dst * D + c);
  asm volatile("global_atomic_pk_add_bf16 %0, %1, off" :: "v"(addr),        "v"(d.x) : "memory");
  asm volatile("global_atomic_pk_add_bf16 %0, %1, off" :: "v"(addr + 4ull), "v"(d.y) : "memory");
}

// ---------------- bf16 WMMA dual-GEMM, fused bias/BN/ReLU ----------------
//
// out = A1 @ B1t^T + A2 @ B2t^T + bias  [+ BN(eval) + ReLU]
// A1,A2: [M,K] bf16 (A1 is pre-scaled aggregate).  B1t,B2t: [NPad,K] bf16.
// 128x128 block tile, 8 waves x (16 rows x 128 cols), K-step 32.
// LDS double-buffered; tiles staged with async global->LDS b64 copies.
__global__ __launch_bounds__(256)
void k_gemm_dual_bf16(const u16* __restrict__ A1, const u16* __restrict__ A2,
                      const u16* __restrict__ B1t, const u16* __restrict__ B2t,
                      const float* __restrict__ bias,
                      const float* __restrict__ bn_g, const float* __restrict__ bn_b,
                      const float* __restrict__ bn_m, const float* __restrict__ bn_v,
                      float* __restrict__ out, u16* __restrict__ out_bf,
                      int M, int K, int Ncols, int NPad, int ldout, int bnrelu)
{
  __shared__ __align__(16) u16 As[2][128][40];  // [buf][row][k], 80B rows
  __shared__ __align__(16) u16 Bt[2][128][40];  // [buf][n][k]

  const int tid  = threadIdx.x;
  const int wave = tid >> 5;
  const int lane = tid & 31;
  const int mbase = blockIdx.x * 128;
  const int nbase = blockIdx.y * 128;
  const int KT = K >> 5;       // K-tiles per pass
  const int TT = KT << 1;      // two passes (A1@B1, A2@B2)

  v8f acc[8] = {};

  // Stage one 128x32 A tile + 128x32 Bt tile into LDS buffer `buf` using
  // async copies (no VGPR round-trip). Out-of-range rows/cols are clamped:
  // they only feed accumulators the epilogue discards.
  auto stage = [&](int t, int buf) {
    const bool p2 = (t >= KT);
    const u16* __restrict__ A = p2 ? A2 : A1;
    const u16* __restrict__ B = p2 ? B2t : B1t;
    const int k0 = (p2 ? (t - KT) : t) << 5;
    #pragma unroll
    for (int j = 0; j < 4; ++j) {
      int q  = tid + j * 256;
      int r  = q >> 3;
      int c4 = (q & 7) << 2;
      int gr = mbase + r; gr = (gr < M) ? gr : (M - 1);
      async_b64(A + gr * K + k0 + c4,
                (unsigned)(unsigned long long)&As[buf][r][c4]);
    }
    #pragma unroll
    for (int j = 0; j < 4; ++j) {
      int q  = tid + j * 256;
      int n  = q >> 3;
      int c4 = (q & 7) << 2;
      int gn = nbase + n; gn = (gn < NPad) ? gn : (NPad - 1);
      async_b64(B + gn * K + k0 + c4,
                (unsigned)(unsigned long long)&Bt[buf][n][c4]);
    }
  };

  stage(0, 0);
  for (int t = 0; t < TT; ++t) {
    const int cur = t & 1;
    wait_async0();                         // my stage(t) copies done writing LDS
    __syncthreads();                       // everyone's stage(t) visible
    if (t + 1 < TT) stage(t + 1, cur ^ 1); // async next tile overlaps WMMA

    // A fragment (16-bit A 16x32 layout): lane<16 -> K 0..7 & 16..23; lane>=16 -> +8
    const int arow = wave * 16 + (lane & 15);
    const int kh   = (lane >> 4) * 8;
    const int kb   = (lane >> 4) * 16;
    const int bc   = lane & 15;
    union F { v16bf v; uint4 q[2]; };
    F af;
    af.q[0] = *(const uint4*)&As[cur][arow][kh];
    af.q[1] = *(const uint4*)&As[cur][arow][kh + 16];
    // software-pipelined B fragments: load ct+1 before issuing wmma ct
    F bcur, bnxt;
    bcur.q[0] = *(const uint4*)&Bt[cur][bc][kh];
    bcur.q[1] = *(const uint4*)&Bt[cur][bc][kh + 16];
    #pragma unroll
    for (int ct = 0; ct < 8; ++ct) {
      if (ct < 7) {
        bnxt.q[0] = *(const uint4*)&Bt[cur][(ct + 1) * 16 + bc][kh];
        bnxt.q[1] = *(const uint4*)&Bt[cur][(ct + 1) * 16 + bc][kh + 16];
      }
      acc[ct] = __builtin_amdgcn_wmma_f32_16x16x32_bf16(
          false, af.v, false, bcur.v, (short)0, acc[ct], false, false);
      bcur = bnxt;
    }
  }

  // epilogue: bias + BN(eval) + ReLU; write f32 (+ optional bf16 copy for next layer)
  const bool wbf = (out_bf != nullptr);
  const int rsel = (lane >> 4) * 8;        // C/D layout: lanes 16-31 hold rows 8..15
  #pragma unroll
  for (int ct = 0; ct < 8; ++ct) {
    int col = nbase + ct * 16 + (lane & 15);
    if (col >= Ncols) continue;
    float b  = bias[col];
    float sc = 1.0f, sh = 0.0f;
    if (bnrelu) {
      float inv = rsqrtf(bn_v[col] + BN_EPS);
      sc = bn_g[col] * inv;
      sh = bn_b[col] - bn_m[col] * sc;
    }
    #pragma unroll
    for (int r = 0; r < 8; ++r) {
      int row = mbase + wave * 16 + rsel + r;
      if (row >= M) continue;
      float v = acc[ct][r] + b;
      if (bnrelu) v = fmaxf(v * sc + sh, 0.0f);
      out[row * ldout + col] = v;
      if (wbf) out_bf[row * ldout + col] = f2bf(v);
    }
  }
}

// ---------------- log-softmax over 47 cols, one wave per row ----------------
__global__ void k_logsoftmax(const float* __restrict__ z, float* __restrict__ y) {
  int row  = blockIdx.x * (blockDim.x >> 5) + (threadIdx.x >> 5);
  int lane = threadIdx.x & 31;
  if (row >= N_NODES) return;
  const float* zr = z + row * D_OUT;
  float v0 = (lane      < D_OUT) ? zr[lane]      : -INFINITY;
  float v1 = (lane + 32 < D_OUT) ? zr[lane + 32] : -INFINITY;
  float m = fmaxf(v0, v1);
  #pragma unroll
  for (int o = 16; o; o >>= 1) m = fmaxf(m, __shfl_xor(m, o, 32));
  float s = ((lane      < D_OUT) ? __expf(v0 - m) : 0.0f) +
            ((lane + 32 < D_OUT) ? __expf(v1 - m) : 0.0f);
  #pragma unroll
  for (int o = 16; o; o >>= 1) s += __shfl_xor(s, o, 32);
  float ls = __logf(s);
  if (lane < D_OUT)      y[row * D_OUT + lane]      = v0 - m - ls;
  if (lane + 32 < D_OUT) y[row * D_OUT + lane + 32] = v1 - m - ls;
}

// ---------------- host launch ----------------

extern "C" void kernel_launch(void* const* d_in, const int* in_sizes, int n_in,
                              void* d_out, int out_size, void* d_ws, size_t ws_size,
                              hipStream_t stream) {
  const float* x    = (const float*)d_in[0];
  const int*   ei   = (const int*)  d_in[1];
  const float* W1_l = (const float*)d_in[2];
  const float* b1   = (const float*)d_in[3];
  const float* W1_r = (const float*)d_in[4];
  const float* g1   = (const float*)d_in[5];
  const float* bt1  = (const float*)d_in[6];
  const float* mn1  = (const float*)d_in[7];
  const float* vr1  = (const float*)d_in[8];
  const float* W2_l = (const float*)d_in[9];
  const float* b2   = (const float*)d_in[10];
  const float* W2_r = (const float*)d_in[11];
  const float* g2   = (const float*)d_in[12];
  const float* bt2  = (const float*)d_in[13];
  const float* mn2  = (const float*)d_in[14];
  const float* vr2  = (const float*)d_in[15];
  const float* W3_l = (const float*)d_in[16];
  const float* b3   = (const float*)d_in[17];
  const float* W3_r = (const float*)d_in[18];

  float* outp  = (float*)d_out;
  float* z     = outp;                              // [N,47]
  float* ypred = z + (size_t)N_NODES * D_OUT;       // [N,47]
  float* S1    = ypred + (size_t)N_NODES * D_OUT;   // [N,256]
  float* S2    = S1 + (size_t)N_NODES * D_HID;      // [N,256]

  // workspace layout (all 16B-aligned)
  float* inv_deg = (float*)d_ws;                          // N f32
  u16*   agg  = (u16*)(inv_deg + 50176);                  // N*256 bf16 (reused per layer)
  u16*   xb   = agg  + (size_t)N_NODES * D_HID;           // N*128 bf16
  u16*   S1b  = xb   + (size_t)N_NODES * D_IN;            // N*256 bf16
  u16*   S2b  = S1b  + (size_t)N_NODES * D_HID;           // N*256 bf16
  u16*   Wt1l = S2b  + (size_t)N_NODES * D_HID;           // [256][128]
  u16*   Wt1r = Wt1l + 256 * 128;
  u16*   Wt2l = Wt1r + 256 * 128;                         // [256][256]
  u16*   Wt2r = Wt2l + 256 * 256;
  u16*   Wt3l = Wt2r + 256 * 256;                         // [64][256] (47 padded to 64)
  u16*   Wt3r = Wt3l + 64 * 256;

  const int T = 256;

  // degrees -> inv_deg
  k_zero_u32<<<(N_NODES + T - 1) / T, T, 0, stream>>>((unsigned*)inv_deg, N_NODES);
  k_degree  <<<(N_EDGES + T - 1) / T, T, 0, stream>>>(ei, inv_deg);
  k_invdeg  <<<(N_NODES + T - 1) / T, T, 0, stream>>>(inv_deg);

  // one-time per-launch conversions: x -> bf16, weights -> transposed bf16
  k_cvt_bf16_v4<<<(N_NODES * D_IN / 4 + T - 1) / T, T, 0, stream>>>(x, xb, N_NODES * D_IN / 4);
  k_wt_bf16<<<(256 * 128 + T - 1) / T, T, 0, stream>>>(W1_l, Wt1l, 7, 256, 256 * 128);
  k_wt_bf16<<<(256 * 128 + T - 1) / T, T, 0, stream>>>(W1_r, Wt1r, 7, 256, 256 * 128);
  k_wt_bf16<<<(256 * 256 + T - 1) / T, T, 0, stream>>>(W2_l, Wt2l, 8, 256, 256 * 256);
  k_wt_bf16<<<(256 * 256 + T - 1) / T, T, 0, stream>>>(W2_r, Wt2r, 8, 256, 256 * 256);
  k_wt_bf16<<<( 64 * 256 + T - 1) / T, T, 0, stream>>>(W3_l, Wt3l, 8,  47,  64 * 256);
  k_wt_bf16<<<( 64 * 256 + T - 1) / T, T, 0, stream>>>(W3_r, Wt3r, 8,  47,  64 * 256);

  dim3 gemm_grid2((N_NODES + 127) / 128, 2);  // 256 cols
  dim3 gemm_grid3((N_NODES + 127) / 128, 1);  // 47 cols

  // ---- layer 1: agg = sum(inv_deg[dst]*x[src]) ; S1 = relu(bn1(agg@W1_l + b1 + x@W1_r))
  k_zero_u32   <<<(N_NODES * D_IN / 2 + T - 1) / T, T, 0, stream>>>((unsigned*)agg, N_NODES * D_IN / 2);
  k_scatter_bf16<<<N_EDGES / 8, T, 0, stream>>>(xb, ei, inv_deg, agg, D_IN, 5);
  k_gemm_dual_bf16<<<gemm_grid2, T, 0, stream>>>(agg, xb, Wt1l, Wt1r, b1,
                                                 g1, bt1, mn1, vr1,
                                                 S1, S1b, N_NODES, D_IN, D_HID, D_HID, D_HID, 1);

  // ---- layer 2
  k_zero_u32   <<<(N_NODES * D_HID / 2 + T - 1) / T, T, 0, stream>>>((unsigned*)agg, N_NODES * D_HID / 2);
  k_scatter_bf16<<<N_EDGES / 4, T, 0, stream>>>(S1b, ei, inv_deg, agg, D_HID, 6);
  k_gemm_dual_bf16<<<gemm_grid2, T, 0, stream>>>(agg, S1b, Wt2l, Wt2r, b2,
                                                 g2, bt2, mn2, vr2,
                                                 S2, S2b, N_NODES, D_HID, D_HID, D_HID, D_HID, 1);

  // ---- layer 3 (no BN/ReLU): z = agg@W3_l + b3 + S2@W3_r
  k_zero_u32   <<<(N_NODES * D_HID / 2 + T - 1) / T, T, 0, stream>>>((unsigned*)agg, N_NODES * D_HID / 2);
  k_scatter_bf16<<<N_EDGES / 4, T, 0, stream>>>(S2b, ei, inv_deg, agg, D_HID, 6);
  k_gemm_dual_bf16<<<gemm_grid3, T, 0, stream>>>(agg, S2b, Wt3l, Wt3r, b3,
                                                 nullptr, nullptr, nullptr, nullptr,
                                                 z, nullptr, N_NODES, D_HID, D_OUT, 64, D_OUT, 0);

  // ---- log_softmax
  k_logsoftmax<<<(N_NODES + 7) / 8, T, 0, stream>>>(z, ypred);
}
